// BertLinearAttention_12532714570430
// MI455X (gfx1250) — matrix operand order repaired and checked by reading, not compiled, not run on hardware
//
#include <hip/hip_runtime.h>
#include <hip/hip_bf16.h>

typedef __attribute__((ext_vector_type(16))) _Float16 v16h;
typedef __attribute__((ext_vector_type(8)))  float    v8f;
typedef int v4i_ __attribute__((vector_size(16)));   // pointee type of async-LDS builtin

static constexpr int S_   = 4096;
static constexpr int HID_ = 1024;
static constexpr int H_   = 16;
static constexpr int D_   = 64;
static constexpr float SCALE_  = 0.125f;   // 64^-0.5
static constexpr float NEGINF_ = -1e30f;

union F16x16 { v16h v; uint4 q[2]; };
union H8pack { _Float16 h[8]; uint4 u; };

__device__ __forceinline__ v8f wmma_f16(v16h a, v16h b, v8f c) {
    // (neg_a, A, neg_b, B, c_mod, C, reuse_a, reuse_b)
    return __builtin_amdgcn_wmma_f32_16x16x32_f16(false, a, false, b, (short)0, c, false, false);
}

// ---------------------------------------------------------------------------
// gfx1250 async direct-to-LDS copies (ASYNCcnt-tracked), guarded fallback.
// ---------------------------------------------------------------------------
#if __has_builtin(__builtin_amdgcn_global_load_async_to_lds_b128)
#define USE_ASYNC_LDS 1
#else
#define USE_ASYNC_LDS 0
#endif

#if USE_ASYNC_LDS
__device__ __forceinline__ void async_cp16(void* lds_dst, const void* gsrc) {
    __builtin_amdgcn_global_load_async_to_lds_b128(
        (__attribute__((address_space(1))) v4i_*)gsrc,
        (__attribute__((address_space(3))) v4i_*)lds_dst,
        /*imm offset*/ 0, /*cpol*/ 0);
}
__device__ __forceinline__ void async_wait0() {
#if __has_builtin(__builtin_amdgcn_s_wait_asynccnt)
    __builtin_amdgcn_s_wait_asynccnt(0);
#else
    asm volatile("s_wait_asynccnt 0x0" ::: "memory");
#endif
}
#endif

// ---------------------------------------------------------------------------
// f32 -> f16 elementwise convert (4 elems / thread)
// ---------------------------------------------------------------------------
__global__ __launch_bounds__(256) void cvt_f16_kernel(const float* __restrict__ in,
                                                      _Float16* __restrict__ out) {
    size_t i = ((size_t)blockIdx.x * 256 + threadIdx.x) * 4;
    float4 v = *(const float4*)(in + i);
    union { _Float16 h[4]; uint2 u; } p;
    p.h[0] = (_Float16)v.x; p.h[1] = (_Float16)v.y;
    p.h[2] = (_Float16)v.z; p.h[3] = (_Float16)v.w;
    *(uint2*)(out + i) = p.u;
}

// ---------------------------------------------------------------------------
// W [in=1024][out=1024] f32  ->  Wt [out][in] f16  (LDS-tiled transpose)
// ---------------------------------------------------------------------------
__global__ __launch_bounds__(256) void transpose_cvt_kernel(const float* __restrict__ in,
                                                            _Float16* __restrict__ out) {
    __shared__ float tile[32][33];
    int x  = blockIdx.x * 32 + threadIdx.x;
    int y0 = blockIdx.y * 32;
    #pragma unroll
    for (int j = threadIdx.y; j < 32; j += 8)
        tile[j][threadIdx.x] = in[(size_t)(y0 + j) * HID_ + x];
    __syncthreads();
    int ox  = blockIdx.y * 32 + threadIdx.x;
    int oy0 = blockIdx.x * 32;
    #pragma unroll
    for (int j = threadIdx.y; j < 32; j += 8)
        out[(size_t)(oy0 + j) * HID_ + ox] = (_Float16)tile[threadIdx.x][j];
}

// ---------------------------------------------------------------------------
// QKV projection GEMM: y = x @ W + b, double-buffered LDS, WMMA f16->f32.
// Block = 8 waves, tile 128 rows x 64 cols. blockIdx.z selects Q/K/V.
// Q,K stored [H][S][D] f16; V stored transposed [H][D][S] f16.
// ---------------------------------------------------------------------------
__global__ __launch_bounds__(256) void qkv_gemm_kernel(
    const _Float16* __restrict__ x16,
    const _Float16* __restrict__ wqT, const _Float16* __restrict__ wkT,
    const _Float16* __restrict__ wvT,
    const float* __restrict__ bq, const float* __restrict__ bk,
    const float* __restrict__ bv,
    _Float16* __restrict__ q16, _Float16* __restrict__ k16,
    _Float16* __restrict__ v16T)
{
    const int which = blockIdx.z;
    const _Float16* wT   = (which == 0) ? wqT : (which == 1) ? wkT : wvT;
    const float*    bias = (which == 0) ? bq : (which == 1) ? bk : bv;

    const int r0 = blockIdx.x * 128;
    const int n0 = blockIdx.y * 64;

    __shared__ __align__(16) _Float16 A_lds[2][128][32];   // x rows x k
    __shared__ __align__(16) _Float16 B_lds[2][64][32];    // Wt rows (n) x k

    const int t     = threadIdx.x;
    const int wave  = t >> 5;
    const int lane  = t & 31;
    const int lhalf = lane >> 4;
    const int l16   = lane & 15;

    // staging addresses (per thread: 32B of A, 16B of B per tile)
    const int arow = t >> 1;
    const int aoff = (t & 1) * 16;
    const int bn   = t >> 2;
    const int bseg = (t & 3) * 8;
    const _Float16* aptr = x16 + (size_t)(r0 + arow) * HID_ + aoff;
    const _Float16* bptr = wT  + (size_t)(n0 + bn)   * HID_ + bseg;

    v8f acc[4] = {};

#if USE_ASYNC_LDS
    auto stage = [&](int buf, int k0) {
        async_cp16(&A_lds[buf][arow][aoff],     aptr + k0);
        async_cp16(&A_lds[buf][arow][aoff + 8], aptr + k0 + 8);
        async_cp16(&B_lds[buf][bn][bseg],       bptr + k0);
    };
    stage(0, 0);
    async_wait0();
    __syncthreads();
#else
    uint4 ra0, ra1, rb;
    auto fetch = [&](int k0) {
        ra0 = *(const uint4*)(aptr + k0);
        ra1 = *(const uint4*)(aptr + k0 + 8);
        rb  = *(const uint4*)(bptr + k0);
    };
    auto commit = [&](int buf) {
        *(uint4*)&A_lds[buf][arow][aoff]     = ra0;
        *(uint4*)&A_lds[buf][arow][aoff + 8] = ra1;
        *(uint4*)&B_lds[buf][bn][bseg]       = rb;
    };
    fetch(0); commit(0);
    __syncthreads();
#endif

    for (int k0 = 0; k0 < HID_; k0 += 32) {
        const int  cur  = (k0 >> 5) & 1;
        const int  nxt  = cur ^ 1;
        const bool more = (k0 + 32) < HID_;

#if USE_ASYNC_LDS
        if (more) stage(nxt, k0 + 32);       // async DMA overlaps the WMMAs below
#else
        if (more) fetch(k0 + 32);            // loads in flight during WMMAs
#endif

        F16x16 af;
        af.q[0] = *(const uint4*)&A_lds[cur][wave * 16 + l16][lhalf * 16];
        af.q[1] = *(const uint4*)&A_lds[cur][wave * 16 + l16][lhalf * 16 + 8];
        #pragma unroll
        for (int j = 0; j < 4; ++j) {
            F16x16 bf;
            const uint4* bp = (const uint4*)&B_lds[cur][j * 16 + l16][lhalf * 16];
            bf.q[0] = bp[0]; bf.q[1] = bp[1];
            acc[j] = wmma_f16(af.v, bf.v, acc[j]);
        }

#if USE_ASYNC_LDS
        if (more) async_wait0();
#else
        if (more) commit(nxt);
#endif
        __syncthreads();
    }

    // epilogue: bias add + store
    #pragma unroll
    for (int j = 0; j < 4; ++j) {
        int ng = n0 + j * 16 + l16;
        float b = bias[ng];
        int h = ng >> 6, d = ng & 63;
        if (which < 2) {
            _Float16* out = (which == 0) ? q16 : k16;
            #pragma unroll
            for (int r = 0; r < 8; ++r) {
                int m = r0 + wave * 16 + r + 8 * lhalf;
                out[((size_t)h * S_ + m) * D_ + d] = (_Float16)(acc[j][r] + b);
            }
        } else {
            H8pack p;
            #pragma unroll
            for (int r = 0; r < 8; ++r) p.h[r] = (_Float16)(acc[j][r] + b);
            *(uint4*)&v16T[(size_t)ng * S_ + r0 + wave * 16 + 8 * lhalf] = p.u;
        }
    }
}

// ---------------------------------------------------------------------------
// Flash attention: block = (head, 128 q-rows), 8 waves, 64-key chunks,
// double-buffered K/V LDS (async staging when available).
// ---------------------------------------------------------------------------
__global__ __launch_bounds__(256) void attention_kernel(
    const _Float16* __restrict__ q16, const _Float16* __restrict__ k16,
    const _Float16* __restrict__ v16T, const int* __restrict__ amask,
    float* __restrict__ out)
{
    const int h  = blockIdx.y;
    const int q0 = blockIdx.x * 128;
    const int t     = threadIdx.x;
    const int wave  = t >> 5;
    const int lane  = t & 31;
    const int lhalf = lane >> 4;
    const int l16   = lane & 15;

    __shared__ __align__(16) _Float16 K_lds[2][64][64];    // [key][d]
    __shared__ __align__(16) _Float16 V_lds[2][64][64];    // [d][key]
    __shared__ __align__(16) _Float16 P_lds[8][16][64];    // per-wave P scratch

    // staging: 32B K + 32B V per thread per chunk
    const int srow = t >> 2;
    const int sseg = (t & 3) * 16;
    const _Float16* kptr = k16  + ((size_t)h * S_ + srow) * D_ + sseg; // + kc*D_
    const _Float16* vptr = v16T + ((size_t)h * D_ + srow) * S_ + sseg; // + kc

    // preload Q A-fragments (16 rows x 64 d = two 16x32 fragments)
    const _Float16* qbase = q16 + ((size_t)h * S_ + q0 + wave * 16 + l16) * D_;
    F16x16 qa[2];
    #pragma unroll
    for (int c = 0; c < 2; ++c) {
        qa[c].q[0] = *(const uint4*)(qbase + c * 32 + lhalf * 16);
        qa[c].q[1] = *(const uint4*)(qbase + c * 32 + lhalf * 16 + 8);
    }

#if USE_ASYNC_LDS
    auto stage = [&](int buf, int kc) {
        async_cp16(&K_lds[buf][srow][sseg],     kptr + (size_t)kc * D_);
        async_cp16(&K_lds[buf][srow][sseg + 8], kptr + (size_t)kc * D_ + 8);
        async_cp16(&V_lds[buf][srow][sseg],     vptr + kc);
        async_cp16(&V_lds[buf][srow][sseg + 8], vptr + kc + 8);
    };
    stage(0, 0);
    async_wait0();
    __syncthreads();
#else
    uint4 rk0, rk1, rv0, rv1;
    auto fetch = [&](int kc) {
        rk0 = *(const uint4*)(kptr + (size_t)kc * D_);
        rk1 = *(const uint4*)(kptr + (size_t)kc * D_ + 8);
        rv0 = *(const uint4*)(vptr + kc);
        rv1 = *(const uint4*)(vptr + kc + 8);
    };
    auto commit = [&](int buf) {
        *(uint4*)&K_lds[buf][srow][sseg]     = rk0;
        *(uint4*)&K_lds[buf][srow][sseg + 8] = rk1;
        *(uint4*)&V_lds[buf][srow][sseg]     = rv0;
        *(uint4*)&V_lds[buf][srow][sseg + 8] = rv1;
    };
    fetch(0); commit(0);
    __syncthreads();
#endif

    v8f o[4] = {};
    float mrun[8], lrun[8];
    #pragma unroll
    for (int r = 0; r < 8; ++r) { mrun[r] = -3.0e38f; lrun[r] = 0.0f; }

    for (int kc = 0; kc < S_; kc += 64) {
        const int  cur  = (kc >> 6) & 1;
        const int  nxt  = cur ^ 1;
        const bool more = (kc + 64) < S_;

#if USE_ASYNC_LDS
        if (more) stage(nxt, kc + 64);
#else
        if (more) fetch(kc + 64);
#endif

        // scores: four 16x16 key tiles, 8 WMMAs
        v8f s[4] = {};
        #pragma unroll
        for (int c = 0; c < 2; ++c) {
            #pragma unroll
            for (int jt = 0; jt < 4; ++jt) {
                F16x16 bf;
                const uint4* p = (const uint4*)&K_lds[cur][jt * 16 + l16][c * 32 + lhalf * 16];
                bf.q[0] = p[0]; bf.q[1] = p[1];
                s[jt] = wmma_f16(qa[c].v, bf.v, s[jt]);
            }
        }

        // scale + key mask (one key per lane in C layout)
        bool ok[4];
        #pragma unroll
        for (int jt = 0; jt < 4; ++jt) ok[jt] = amask[kc + jt * 16 + l16] != 0;
        #pragma unroll
        for (int jt = 0; jt < 4; ++jt)
            #pragma unroll
            for (int r = 0; r < 8; ++r)
                s[jt][r] = ok[jt] ? s[jt][r] * SCALE_ : NEGINF_;

        // online softmax (row reductions across the 16 lanes of each half)
        float alpha[8];
        #pragma unroll
        for (int r = 0; r < 8; ++r) {
            float tmax = fmaxf(fmaxf(s[0][r], s[1][r]), fmaxf(s[2][r], s[3][r]));
            #pragma unroll
            for (int mk = 1; mk < 16; mk <<= 1)
                tmax = fmaxf(tmax, __shfl_xor(tmax, mk, 32));
            float mnew = fmaxf(mrun[r], tmax);
            alpha[r] = __expf(mrun[r] - mnew);
            mrun[r] = mnew;
            float rsum = 0.0f;
            #pragma unroll
            for (int jt = 0; jt < 4; ++jt) {
                s[jt][r] = __expf(s[jt][r] - mnew);
                rsum += s[jt][r];
            }
            #pragma unroll
            for (int mk = 1; mk < 16; mk <<= 1)
                rsum += __shfl_xor(rsum, mk, 32);
            lrun[r] = lrun[r] * alpha[r] + rsum;
        }
        #pragma unroll
        for (int j = 0; j < 4; ++j)
            #pragma unroll
            for (int r = 0; r < 8; ++r)
                o[j][r] *= alpha[r];

        // P: C layout -> per-wave LDS -> A layout (same-wave DS ops in-order)
        #pragma unroll
        for (int r = 0; r < 8; ++r) {
            int m = r + 8 * lhalf;
            #pragma unroll
            for (int jt = 0; jt < 4; ++jt)
                P_lds[wave][m][jt * 16 + l16] = (_Float16)s[jt][r];
        }
        asm volatile("" ::: "memory");

        // O += P x V : two 32-deep key halves x four d tiles = 8 WMMAs
        #pragma unroll
        for (int ka = 0; ka < 2; ++ka) {
            F16x16 pa;
            pa.q[0] = *(const uint4*)&P_lds[wave][l16][ka * 32 + lhalf * 16];
            pa.q[1] = *(const uint4*)&P_lds[wave][l16][ka * 32 + lhalf * 16 + 8];
            #pragma unroll
            for (int j = 0; j < 4; ++j) {
                F16x16 vb;
                const uint4* vp = (const uint4*)&V_lds[cur][j * 16 + l16][ka * 32 + lhalf * 16];
                vb.q[0] = vp[0]; vb.q[1] = vp[1];
                o[j] = wmma_f16(pa.v, vb.v, o[j]);
            }
        }

#if USE_ASYNC_LDS
        if (more) async_wait0();
#else
        if (more) commit(nxt);
#endif
        __syncthreads();
    }

    // normalize + store ctx[q][h*64+d] f32
    #pragma unroll
    for (int j = 0; j < 4; ++j) {
        int d = j * 16 + l16;
        #pragma unroll
        for (int r = 0; r < 8; ++r) {
            int m = q0 + wave * 16 + r + 8 * lhalf;
            out[(size_t)m * HID_ + h * D_ + d] = o[j][r] / lrun[r];
        }
    }
}

// ---------------------------------------------------------------------------
// Host-side launch
// ---------------------------------------------------------------------------
extern "C" void kernel_launch(void* const* d_in, const int* in_sizes, int n_in,
                              void* d_out, int out_size, void* d_ws, size_t ws_size,
                              hipStream_t stream) {
    const float* x    = (const float*)d_in[0];
    const int*   mask = (const int*)d_in[1];
    const float* Wq   = (const float*)d_in[2];
    const float* bq   = (const float*)d_in[3];
    const float* Wk   = (const float*)d_in[4];
    const float* bk   = (const float*)d_in[5];
    const float* Wv   = (const float*)d_in[6];
    const float* bv   = (const float*)d_in[7];
    float* out = (float*)d_out;

    char* ws = (char*)d_ws;
    _Float16* x16  = (_Float16*)(ws);                          //  0 ..  8MB
    _Float16* wqT  = (_Float16*)(ws + (size_t)(8  << 20));     //  8 .. 10MB
    _Float16* wkT  = (_Float16*)(ws + (size_t)(10 << 20));     // 10 .. 12MB
    _Float16* wvT  = (_Float16*)(ws + (size_t)(12 << 20));     // 12 .. 14MB
    _Float16* q16  = (_Float16*)(ws + (size_t)(14 << 20));     // 14 .. 22MB
    _Float16* k16  = (_Float16*)(ws + (size_t)(22 << 20));     // 22 .. 30MB
    _Float16* v16T = (_Float16*)(ws + (size_t)(30 << 20));     // 30 .. 38MB

    cvt_f16_kernel<<<(S_ * HID_) / (256 * 4), 256, 0, stream>>>(x, x16);

    dim3 tb(32, 8), tg(HID_ / 32, HID_ / 32);
    transpose_cvt_kernel<<<tg, tb, 0, stream>>>(Wq, wqT);
    transpose_cvt_kernel<<<tg, tb, 0, stream>>>(Wk, wkT);
    transpose_cvt_kernel<<<tg, tb, 0, stream>>>(Wv, wvT);

    qkv_gemm_kernel<<<dim3(S_ / 128, HID_ / 64, 3), 256, 0, stream>>>(
        x16, wqT, wkT, wvT, bq, bk, bv, q16, k16, v16T);

    attention_kernel<<<dim3(S_ / 128, H_), 256, 0, stream>>>(
        q16, k16, v16T, mask, out);
}
